// PointContext3D_249108103884
// MI455X (gfx1250) — matrix operand order repaired and checked by reading, not compiled, not run on hardware
//
#include <hip/hip_runtime.h>

#define NPTS 3072
#define BATCH 2
#define NB (BATCH * NPTS)   // 6144 columns (b,n)
#define CDIM 128
#define INDIM 512
#define GRP 4
#define CGRP 32
#define EPSV 1e-5f
#define NCHUNK (NPTS / 32)  // 96 key chunks in flash attention

#define USE_ASYNC_COPY 1

typedef __attribute__((ext_vector_type(16))) _Float16 v16h;
typedef __attribute__((ext_vector_type(8)))  _Float16 v8h;
typedef __attribute__((ext_vector_type(4)))  _Float16 v4h;
typedef __attribute__((ext_vector_type(8)))  float    v8f;

static __device__ inline v8f wmma_f16(v16h a, v16h b, v8f c) {
  return __builtin_amdgcn_wmma_f32_16x16x32_f16(false, a, false, b, (short)0, c, false, false);
}

// A fragment (16x32 f16, M x K): lane m = lane%16; k = h + 16*(lane>=16).
static __device__ inline v16h load_frag_A(const _Float16* __restrict__ row, int hi) {
  const _Float16* p = row + (hi ? 16 : 0);
  v8h lo = *(const v8h*)(p);
  v8h hh = *(const v8h*)(p + 8);
  v16h f;
#pragma unroll
  for (int h = 0; h < 8; ++h) { f[h] = lo[h]; f[h + 8] = hh[h]; }
  return f;
}

// B fragment (32x16 f16, K x N): lane n = lane%16; k = h + 16*(lane>=16).
static __device__ inline v16h load_frag_B(const _Float16* __restrict__ row, int hi) {
  const _Float16* p = row + (hi ? 16 : 0);
  v8h lo = *(const v8h*)(p);
  v8h hh = *(const v8h*)(p + 8);
  v16h f;
#pragma unroll
  for (int h = 0; h < 8; ++h) { f[h] = lo[h]; f[h + 8] = hh[h]; }
  return f;
}

#if USE_ASYNC_COPY
// CDNA5 async global->LDS copies (ASYNCcnt-tracked). LDS dest is the 32-bit LDS
// byte offset (addrspace(3) ptrtoint); global addr is a 64-bit VGPR pair.
#define LDS_OFF(p) ((unsigned)(uintptr_t)(__attribute__((address_space(3))) void*)(p))
static __device__ inline void async_copy_b64(unsigned lds_off, const void* g) {
  asm volatile("global_load_async_to_lds_b64 %0, %1, off"
               :: "v"(lds_off), "v"((unsigned long long)(uintptr_t)g) : "memory");
}
static __device__ inline void async_copy_b128(unsigned lds_off, const void* g) {
  asm volatile("global_load_async_to_lds_b128 %0, %1, off"
               :: "v"(lds_off), "v"((unsigned long long)(uintptr_t)g) : "memory");
}
static __device__ inline void wait_async_le2() {
  asm volatile("s_wait_asynccnt 0x2" ::: "memory");
}
static __device__ inline void wait_async_0() {
  asm volatile("s_wait_asynccnt 0x0" ::: "memory");
}
#endif

// ---------------------------------------------------------------------------
// Generic 128-row WMMA GEMM:  out[m, col] = sum_k W[m,k] * X[k, col] (+bias)
// fp32 inputs converted to f16 while staging in LDS; f32 accumulate.
// Software pipeline: next K-slab's global loads prefetched into registers
// while the current slab's WMMAs run; B fragments batched 4 at a time so
// v_wmma issue back-to-back instead of ds_load->wait->wmma chains.
// outF16T writes the result transposed as [(b*G+h)*N + col][32c] (for flash).
// ---------------------------------------------------------------------------
__global__ __launch_bounds__(256)
void wmma_gemm_kernel(const float* __restrict__ W, const float* __restrict__ X,
                      const float* __restrict__ bias,
                      float* __restrict__ outF32, _Float16* __restrict__ outF16,
                      _Float16* __restrict__ outF16T,
                      int Kdim, int x_row_stride, int x_batch_stride, int cols_per_batch) {
  const int ctile = blockIdx.x * 128;
  const int tid = threadIdx.x;
  const int w  = tid >> 5;
  const int ln = tid & 15;
  const int hi = (tid >> 4) & 1;

  __shared__ __align__(16) _Float16 ldsW[128][40];  // [m][k]
  __shared__ __align__(16) _Float16 ldsX[128][40];  // [n][k] (transposed)

  const int bidx  = ctile / cols_per_batch;           // tiles never straddle a batch
  const int ncol0 = ctile - bidx * cols_per_batch;
  const float* xbase = X + (size_t)bidx * x_batch_stride + ncol0;

  const v8f z8 = {0.f, 0.f, 0.f, 0.f, 0.f, 0.f, 0.f, 0.f};
  v8f acc[8];
#pragma unroll
  for (int nt = 0; nt < 8; ++nt) acc[nt] = z8;

  const int wrow = tid >> 1, wkk = (tid & 1) * 16;    // W stage: 128 rows x 32 k
  const int xk   = tid >> 3, xnb = (tid & 7) * 16;    // X stage: 32 k x 128 cols

  float4 wreg[4], xreg[4];
  {
    const float4* ws = (const float4*)(W + (size_t)wrow * Kdim + wkk);
    const float4* xs = (const float4*)(xbase + (size_t)xk * x_row_stride + xnb);
#pragma unroll
    for (int q = 0; q < 4; ++q) { wreg[q] = ws[q]; xreg[q] = xs[q]; }
  }

  for (int k0 = 0; k0 < Kdim; k0 += 32) {
    // stage current slab (f32 -> f16)
#pragma unroll
    for (int q = 0; q < 4; ++q) {
      ldsW[wrow][wkk + q * 4 + 0] = (_Float16)wreg[q].x;
      ldsW[wrow][wkk + q * 4 + 1] = (_Float16)wreg[q].y;
      ldsW[wrow][wkk + q * 4 + 2] = (_Float16)wreg[q].z;
      ldsW[wrow][wkk + q * 4 + 3] = (_Float16)wreg[q].w;
      ldsX[xnb + q * 4 + 0][xk] = (_Float16)xreg[q].x;
      ldsX[xnb + q * 4 + 1][xk] = (_Float16)xreg[q].y;
      ldsX[xnb + q * 4 + 2][xk] = (_Float16)xreg[q].z;
      ldsX[xnb + q * 4 + 3][xk] = (_Float16)xreg[q].w;
    }
    __syncthreads();
    // prefetch next slab while WMMAs run
    if (k0 + 32 < Kdim) {
      const float4* ws = (const float4*)(W + (size_t)wrow * Kdim + k0 + 32 + wkk);
      const float4* xs = (const float4*)(xbase + (size_t)(k0 + 32 + xk) * x_row_stride + xnb);
#pragma unroll
      for (int q = 0; q < 4; ++q) { wreg[q] = ws[q]; xreg[q] = xs[q]; }
    }
    v16h Af = load_frag_A(&ldsW[w * 16 + ln][0], hi);
#pragma unroll
    for (int g2 = 0; g2 < 2; ++g2) {
      v16h Bf0 = load_frag_B(&ldsX[(g2 * 4 + 0) * 16 + ln][0], hi);
      v16h Bf1 = load_frag_B(&ldsX[(g2 * 4 + 1) * 16 + ln][0], hi);
      v16h Bf2 = load_frag_B(&ldsX[(g2 * 4 + 2) * 16 + ln][0], hi);
      v16h Bf3 = load_frag_B(&ldsX[(g2 * 4 + 3) * 16 + ln][0], hi);
      acc[g2 * 4 + 0] = wmma_f16(Af, Bf0, acc[g2 * 4 + 0]);
      acc[g2 * 4 + 1] = wmma_f16(Af, Bf1, acc[g2 * 4 + 1]);
      acc[g2 * 4 + 2] = wmma_f16(Af, Bf2, acc[g2 * 4 + 2]);
      acc[g2 * 4 + 3] = wmma_f16(Af, Bf3, acc[g2 * 4 + 3]);
    }
    __syncthreads();   // reads done before next slab's stores
  }

  const int m0 = w * 16 + 8 * hi;            // D layout: row = r + 8*(lane>=16)
#pragma unroll
  for (int nt = 0; nt < 8; ++nt) {
    const int col = ctile + nt * 16 + ln;
    if (outF32 || outF16) {
#pragma unroll
      for (int r = 0; r < 8; ++r) {
        const int m = m0 + r;
        float v = acc[nt][r];
        if (bias)   v += bias[m];
        if (outF32) outF32[(size_t)m * NB + col] = v;
        if (outF16) outF16[(size_t)m * NB + col] = (_Float16)v;
      }
    }
    if (outF16T) {   // transposed: [(b*GRP + h)*NPTS + i][32c]; c0 in {0,8,16,24}
      const int b = col / NPTS, i = col - b * NPTS;
      const int h = m0 >> 5, c0 = m0 & 31;
      v8h pack;
#pragma unroll
      for (int r = 0; r < 8; ++r) pack[r] = (_Float16)acc[nt][r];
      *(v8h*)(outF16T + ((size_t)(b * GRP + h) * NPTS + i) * CGRP + c0) = pack;
    }
  }
}

// ---------------------------------------------------------------------------
// Fused flash attention per (b, h):  S^T[j,i] = sum_c G[c,j] P[c,i]
// online softmax over j (within-lane + one shfl_xor(16) pair-merge),
// O[c,i] += T[c,j] * exp(S - m).
// P/G arrive in transposed global layout -> all LDS staging is a pure copy,
// done with CDNA5 async global->LDS loads, double-buffered per key chunk.
// ---------------------------------------------------------------------------
__global__ __launch_bounds__(256)
void flash_attn_kernel(const _Float16* __restrict__ pT, const _Float16* __restrict__ gT,
                       const _Float16* __restrict__ th, float* __restrict__ attn) {
  const int bh = blockIdx.y;
  const int b = bh >> 2, h = bh & 3;
  const int ibase = blockIdx.x * 128;
  const int tid = threadIdx.x;
  const int w  = tid >> 5;
  const int ln = tid & 15;
  const int hi = (tid >> 4) & 1;

  const _Float16* pbh = pT + (size_t)bh * NPTS * CGRP;             // [i][32c]
  const _Float16* gbh = gT + (size_t)bh * NPTS * CGRP;             // [j][32c]
  const size_t trowbase = (size_t)(h * CGRP) * NB + (size_t)b * NPTS;  // th natural [c][col]

  __shared__ __align__(16) _Float16 ldsP[128][40];     // [i][c]
  __shared__ __align__(16) _Float16 ldsG[2][32][40];   // [j][c], double buffered
  __shared__ __align__(16) _Float16 ldsT[2][32][40];   // [c][j], double buffered

  // staging coordinates
  const int pi = tid >> 1, pc = (tid & 1) * 16;        // P: 128 i x 32 c, 16 halves/thread
  const int gj = tid >> 3, gc = (tid & 7) * 4;         // G: 32 j x 32 c, 4 halves/thread
  const int tc = tid >> 3, tj = (tid & 7) * 4;         // T: 32 c x 32 j, 4 halves/thread

#if USE_ASYNC_COPY
  // P tile (once)
  async_copy_b128(LDS_OFF(&ldsP[pi][pc]),     pbh + (size_t)(ibase + pi) * CGRP + pc);
  async_copy_b128(LDS_OFF(&ldsP[pi][pc + 8]), pbh + (size_t)(ibase + pi) * CGRP + pc + 8);
  // chunk 0 -> buf 0
  async_copy_b64(LDS_OFF(&ldsG[0][gj][gc]), gbh + (size_t)gj * CGRP + gc);
  async_copy_b64(LDS_OFF(&ldsT[0][tc][tj]), th + trowbase + (size_t)tc * NB + tj);
#else
  {
    const v8h* s = (const v8h*)(pbh + (size_t)(ibase + pi) * CGRP + pc);
    *(v8h*)&ldsP[pi][pc] = s[0];
    *(v8h*)&ldsP[pi][pc + 8] = s[1];
    *(v4h*)&ldsG[0][gj][gc] = *(const v4h*)(gbh + (size_t)gj * CGRP + gc);
    *(v4h*)&ldsT[0][tc][tj] = *(const v4h*)(th + trowbase + (size_t)tc * NB + tj);
  }
#endif

  const v8f z8 = {0.f, 0.f, 0.f, 0.f, 0.f, 0.f, 0.f, 0.f};
  v8f o0 = z8, o1 = z8;
  float m_run = -3.0e38f, l_run = 0.f;
  v16h Bp;

  auto stage_chunk = [&](int buf, int j0) {
#if USE_ASYNC_COPY
    async_copy_b64(LDS_OFF(&ldsG[buf][gj][gc]), gbh + (size_t)(j0 + gj) * CGRP + gc);
    async_copy_b64(LDS_OFF(&ldsT[buf][tc][tj]), th + trowbase + (size_t)tc * NB + j0 + tj);
#else
    *(v4h*)&ldsG[buf][gj][gc] = *(const v4h*)(gbh + (size_t)(j0 + gj) * CGRP + gc);
    *(v4h*)&ldsT[buf][tc][tj] = *(const v4h*)(th + trowbase + (size_t)tc * NB + j0 + tj);
#endif
  };

  auto compute_chunk = [&](const _Float16 (*Gb)[40], const _Float16 (*Tb)[40]) {
    // scores (transposed): rows j (32), cols i (16)
    v16h Ag0 = load_frag_A(&Gb[ln][0], hi);
    v16h Ag1 = load_frag_A(&Gb[ln + 16][0], hi);
    v8f st0 = wmma_f16(Ag0, Bp, z8);
    v8f st1 = wmma_f16(Ag1, Bp, z8);

    float cm = -3.0e38f;
#pragma unroll
    for (int r = 0; r < 8; ++r) cm = fmaxf(cm, fmaxf(st0[r], st1[r]));
    cm = fmaxf(cm, __shfl_xor(cm, 16));
    const float m_new = fmaxf(m_run, cm);
    const float corr = __expf(m_run - m_new);
    m_run = m_new;
    l_run *= corr;
#pragma unroll
    for (int r = 0; r < 8; ++r) { o0[r] *= corr; o1[r] *= corr; }

    float e0[8], e1[8], lsum = 0.f;
#pragma unroll
    for (int r = 0; r < 8; ++r) {
      e0[r] = __expf(st0[r] - m_new);
      e1[r] = __expf(st1[r] - m_new);
      lsum += e0[r] + e1[r];
    }
    l_run += lsum + __shfl_xor(lsum, 16);

    // D-layout -> B-fragment layout (k=j, n=i) via lane-pair swap
    v16h Bpr;
#pragma unroll
    for (int r = 0; r < 8; ++r) {
      const float f0 = __shfl_xor(e0[r], 16);
      const float f1 = __shfl_xor(e1[r], 16);
      Bpr[r]     = (_Float16)(hi ? f1 : e0[r]);
      Bpr[r + 8] = (_Float16)(hi ? e1[r] : f0);
    }

    v16h At0 = load_frag_A(&Tb[ln][0], hi);        // T rows c=0..15
    v16h At1 = load_frag_A(&Tb[ln + 16][0], hi);   // T rows c=16..31
    o0 = wmma_f16(At0, Bpr, o0);
    o1 = wmma_f16(At1, Bpr, o1);
  };

  // peeled chunk 0: issue chunk 1, wait for P + chunk 0, compute
  stage_chunk(1, 32);
#if USE_ASYNC_COPY
  wait_async_le2();
#endif
  __syncthreads();
  Bp = load_frag_B(&ldsP[w * 16 + ln][0], hi);     // P[c,i], reused for all chunks
  compute_chunk(ldsG[0], ldsT[0]);

  for (int n = 1; n < NCHUNK; ++n) {
    const int cur = n & 1;
    __syncthreads();                                // prev buffer reads complete
    if (n + 1 < NCHUNK) {
      stage_chunk(cur ^ 1, (n + 1) * 32);
#if USE_ASYNC_COPY
      wait_async_le2();                             // in-order => chunk n resident
#endif
    } else {
#if USE_ASYNC_COPY
      wait_async_0();
#endif
    }
    __syncthreads();
    compute_chunk(ldsG[cur], ldsT[cur]);
  }

  const float inv = (l_run > 0.f) ? 1.f / l_run : 0.f;
  const size_t colo = (size_t)b * NPTS + ibase + w * 16 + ln;
#pragma unroll
  for (int r = 0; r < 8; ++r) {
    const int c0 = r + 8 * hi;
    attn[(size_t)(h * CGRP + c0) * NB + colo]      = o0[r] * inv;
    attn[(size_t)(h * CGRP + c0 + 16) * NB + colo] = o1[r] * inv;
  }
}

// ---------------------------------------------------------------------------
// Small kernels: BN stats/apply, grouped z-conv, GroupNorm stats, GN+residual
// ---------------------------------------------------------------------------
__global__ __launch_bounds__(256)
void bn_stats_kernel(const float* __restrict__ y, const float* __restrict__ gamma,
                     const float* __restrict__ beta, float* __restrict__ st) {
  const int ch = blockIdx.x;
  const float* row = y + (size_t)ch * NB;
  float s = 0.f, ss = 0.f;
  for (int i = threadIdx.x; i < NB; i += 256) { float v = row[i]; s += v; ss += v * v; }
  __shared__ float rs[256], rq[256];
  rs[threadIdx.x] = s; rq[threadIdx.x] = ss;
  __syncthreads();
  for (int off = 128; off > 0; off >>= 1) {
    if (threadIdx.x < off) { rs[threadIdx.x] += rs[threadIdx.x + off]; rq[threadIdx.x] += rq[threadIdx.x + off]; }
    __syncthreads();
  }
  if (threadIdx.x == 0) {
    const float m = rs[0] / (float)NB;
    const float var = rq[0] / (float)NB - m * m;
    const float sc = gamma[ch] * rsqrtf(var + EPSV);
    st[2 * ch] = sc;
    st[2 * ch + 1] = beta[ch] - m * sc;
  }
}

__global__ __launch_bounds__(256)
void bn_apply_kernel(const float* __restrict__ y, const float* __restrict__ st,
                     float* __restrict__ xf) {
  const int idx = blockIdx.x * 256 + threadIdx.x;
  if (idx >= CDIM * NB) return;
  const int ch = idx / NB;
  const float v = y[idx] * st[2 * ch] + st[2 * ch + 1];
  xf[idx] = fmaxf(v, 0.f);
}

__global__ __launch_bounds__(256)
void zconv_kernel(const float* __restrict__ wz, const float* __restrict__ attn,
                  float* __restrict__ o2) {
  const int o = blockIdx.y;              // 0..127 ; wz[h][o%32][c] flat = o*CG + c
  const int h = o >> 5;
  const int col = blockIdx.x * 256 + threadIdx.x;
  const float* wr = wz + (size_t)o * CGRP;          // uniform -> scalar loads
  const float* ar = attn + (size_t)(h * CGRP) * NB + col;
  float s = 0.f;
#pragma unroll
  for (int c = 0; c < CGRP; ++c) s = fmaf(wr[c], ar[(size_t)c * NB], s);
  o2[(size_t)o * NB + col] = s;
}

__global__ __launch_bounds__(256)
void gn_stats_kernel(const float* __restrict__ o2, float* __restrict__ st) {
  const int bg = blockIdx.x;             // b*G + g
  const int b = bg >> 2, g = bg & 3;
  const float* base = o2 + (size_t)(g * CGRP) * NB + (size_t)b * NPTS;
  float s = 0.f, ss = 0.f;
  for (int q = threadIdx.x; q < CGRP * NPTS; q += 256) {
    const int c = q / NPTS, n = q - c * NPTS;
    const float v = base[(size_t)c * NB + n];
    s += v; ss += v * v;
  }
  __shared__ float rs[256], rq[256];
  rs[threadIdx.x] = s; rq[threadIdx.x] = ss;
  __syncthreads();
  for (int off = 128; off > 0; off >>= 1) {
    if (threadIdx.x < off) { rs[threadIdx.x] += rs[threadIdx.x + off]; rq[threadIdx.x] += rq[threadIdx.x + off]; }
    __syncthreads();
  }
  if (threadIdx.x == 0) {
    const float cnt = (float)(CGRP * NPTS);
    const float m = rs[0] / cnt;
    const float var = rq[0] / cnt - m * m;
    st[2 * bg] = m;
    st[2 * bg + 1] = rsqrtf(var + EPSV);
  }
}

__global__ __launch_bounds__(256)
void gn_res_kernel(const float* __restrict__ o2, const float* __restrict__ st,
                   const float* __restrict__ gg, const float* __restrict__ gb,
                   const float* __restrict__ x, float* __restrict__ outp, int final_layout) {
  const int idx = blockIdx.x * 256 + threadIdx.x;
  if (idx >= CDIM * NB) return;
  const int ch = idx / NB;
  const int col = idx - ch * NB;
  const int b = col / NPTS;
  const int n = col - b * NPTS;
  const int bg = b * GRP + (ch >> 5);
  const float v = (o2[idx] - st[2 * bg]) * st[2 * bg + 1] * gg[ch] + gb[ch] + x[idx];
  if (final_layout) outp[((size_t)b * CDIM + ch) * NPTS + n] = v;  // [B, C, N]
  else              outp[idx] = v;                                  // [C, NB]
}

// ---------------------------------------------------------------------------
static void run_sa(const float* wt, const float* wp, const float* wg, const float* wz,
                   const float* gng, const float* gnb,
                   float* xf, _Float16* th, _Float16* pT, _Float16* gT,
                   float* attn, float* o2, float* st, float* finalOut, hipStream_t stream) {
  const dim3 blk(256);
  wmma_gemm_kernel<<<dim3(NB / 128), blk, 0, stream>>>(wt, xf, nullptr, nullptr, th, nullptr, CDIM, NB, 0, NB);
  wmma_gemm_kernel<<<dim3(NB / 128), blk, 0, stream>>>(wp, xf, nullptr, nullptr, nullptr, pT, CDIM, NB, 0, NB);
  wmma_gemm_kernel<<<dim3(NB / 128), blk, 0, stream>>>(wg, xf, nullptr, nullptr, nullptr, gT, CDIM, NB, 0, NB);
  flash_attn_kernel<<<dim3(NPTS / 128, BATCH * GRP), blk, 0, stream>>>(pT, gT, th, attn);
  zconv_kernel<<<dim3(NB / 256, CDIM), blk, 0, stream>>>(wz, attn, o2);
  gn_stats_kernel<<<BATCH * GRP, blk, 0, stream>>>(o2, st);
  gn_res_kernel<<<(CDIM * NB) / 256, blk, 0, stream>>>(o2, st, gng, gnb, xf,
                                                       finalOut ? finalOut : xf,
                                                       finalOut != nullptr);
}

extern "C" void kernel_launch(void* const* d_in, const int* in_sizes, int n_in,
                              void* d_out, int out_size, void* d_ws, size_t ws_size,
                              hipStream_t stream) {
  (void)in_sizes; (void)n_in; (void)out_size; (void)ws_size;
  const float* l_features = (const float*)d_in[1];
  const float* conv1_w = (const float*)d_in[3];
  const float* conv1_b = (const float*)d_in[4];
  const float* bn1_g   = (const float*)d_in[5];
  const float* bn1_b   = (const float*)d_in[6];
  const float* conv2_w = (const float*)d_in[7];
  const float* conv2_b = (const float*)d_in[8];
  const float* bn2_g   = (const float*)d_in[9];
  const float* bn2_b   = (const float*)d_in[10];
  const float* sa1_wt  = (const float*)d_in[11];
  const float* sa1_wp  = (const float*)d_in[12];
  const float* sa1_wg  = (const float*)d_in[13];
  const float* sa1_wz  = (const float*)d_in[14];
  const float* sa1_gg  = (const float*)d_in[15];
  const float* sa1_gb  = (const float*)d_in[16];
  const float* sa2_wt  = (const float*)d_in[17];
  const float* sa2_wp  = (const float*)d_in[18];
  const float* sa2_wg  = (const float*)d_in[19];
  const float* sa2_wz  = (const float*)d_in[20];
  const float* sa2_gg  = (const float*)d_in[21];
  const float* sa2_gb  = (const float*)d_in[22];
  float* out = (float*)d_out;

  // workspace carve-out (~16.5 MB)
  const size_t PL = (size_t)CDIM * NB;     // 786432 elements
  float* y    = (float*)d_ws;
  float* xf   = y + PL;
  float* attn = xf + PL;
  float* o2   = attn + PL;
  float* st   = o2 + PL;                   // 256 floats of stats
  _Float16* th = (_Float16*)(st + 256);
  _Float16* pT = th + PL;
  _Float16* gT = pT + PL;

  const dim3 blk(256);

  // conv1 (K=512, batched X = l_features[B,512,N]) + BN1 + ReLU
  wmma_gemm_kernel<<<dim3(NB / 128), blk, 0, stream>>>(conv1_w, l_features, conv1_b, y, nullptr, nullptr,
                                                       INDIM, NPTS, INDIM * NPTS, NPTS);
  bn_stats_kernel<<<CDIM, blk, 0, stream>>>(y, bn1_g, bn1_b, st);
  bn_apply_kernel<<<(CDIM * NB) / 256, blk, 0, stream>>>(y, st, xf);

  // conv2 (K=128, flat X) + BN2 + ReLU
  wmma_gemm_kernel<<<dim3(NB / 128), blk, 0, stream>>>(conv2_w, xf, conv2_b, y, nullptr, nullptr,
                                                       CDIM, NB, 0, NB);
  bn_stats_kernel<<<CDIM, blk, 0, stream>>>(y, bn2_g, bn2_b, st);
  bn_apply_kernel<<<(CDIM * NB) / 256, blk, 0, stream>>>(y, st, xf);

  // SA block 1 (updates xf in place), SA block 2 (writes d_out in [B,C,N] layout)
  run_sa(sa1_wt, sa1_wp, sa1_wg, sa1_wz, sa1_gg, sa1_gb, xf, th, pT, gT, attn, o2, st, nullptr, stream);
  run_sa(sa2_wt, sa2_wp, sa2_wg, sa2_wz, sa2_gg, sa2_gb, xf, th, pT, gT, attn, o2, st, out, stream);
}
